// MultiHeadAttention_54073638256763
// MI455X (gfx1250) — matrix operand order repaired and checked by reading, not compiled, not run on hardware
//
#include <hip/hip_runtime.h>
#include <hip/hip_bf16.h>

// ---------------------------------------------------------------------------
// MHA forward for MI455X (gfx1250, wave32, WMMA bf16 16x16x32).
// Pipeline (all intermediates bf16 in d_ws, L2-resident at ~60MB < 192MB L2):
//   1) cvt_qkv: fp32 Q/K/V -> bf16 row-major
//   2) cvt_w:   fp32 Wq/Wk/Wv/Wo -> bf16 TRANSPOSED (WT[N][K]) for B-operands
//   3) proj_gemm x3: X @ W + b -> q [B,H,S,64], k [B,H,S,64], vT [B,H,64,S]
//   4) attn: flash attention per 16-row q tile (online softmax, causal)
//   5) out_gemm: concat [8192,512] @ Wo + bo -> fp32 d_out
// ---------------------------------------------------------------------------

typedef __bf16 bf16x;
typedef __attribute__((ext_vector_type(16))) __bf16 v16bf;
typedef __attribute__((ext_vector_type(8)))  float  v8f;

union Frag32B {          // 16 bf16 = 32 bytes = 8 VGPRs, loaded as two b128
  v16bf v;
  uint4 q[2];
};

__device__ __forceinline__ v8f vzero8() {
  v8f v;
#pragma unroll
  for (int i = 0; i < 8; ++i) v[i] = 0.0f;
  return v;
}

__device__ __forceinline__ v8f wmma_bf16(v16bf a, v16bf b, v8f c) {
  // D = A(16x32 bf16) x B(32x16 bf16) + C(16x16 f32)
  return __builtin_amdgcn_wmma_f32_16x16x32_bf16(
      /*neg_a=*/false, a, /*neg_b=*/false, b,
      /*c_mod=*/(short)0, c, /*reuse_a=*/false, /*reuse_b=*/false);
}

static constexpr int Bb = 4, Hh = 8, Ss = 2048, Dm = 512, Dk = 64;
static constexpr int Mrows = Bb * Ss;  // 8192

// ---------------------------------------------------------------------------
// 1) fp32 -> bf16 conversion of Q, K, V activations
// ---------------------------------------------------------------------------
__global__ void cvt_qkv_kernel(const float* __restrict__ Q,
                               const float* __restrict__ K,
                               const float* __restrict__ V,
                               bf16x* __restrict__ xq, bf16x* __restrict__ xk,
                               bf16x* __restrict__ xv) {
  size_t i = (size_t)blockIdx.x * blockDim.x + threadIdx.x;
  if (i >= (size_t)Mrows * Dm) return;
  xq[i] = (bf16x)Q[i];
  xk[i] = (bf16x)K[i];
  xv[i] = (bf16x)V[i];
}

// ---------------------------------------------------------------------------
// 2) fp32 W[K][N] -> bf16 WT[N][K] (transposed so B-operand loads are b128s)
// ---------------------------------------------------------------------------
__global__ void cvt_w_kernel(const float* __restrict__ Wq,
                             const float* __restrict__ Wk,
                             const float* __restrict__ Wv,
                             const float* __restrict__ Wo,
                             bf16x* __restrict__ tq, bf16x* __restrict__ tk,
                             bf16x* __restrict__ tv, bf16x* __restrict__ to_) {
  int i = blockIdx.x * blockDim.x + threadIdx.x;
  if (i >= Dm * Dm) return;
  int n = i >> 9;        // output row (N)
  int kk = i & (Dm - 1); // output col (K)
  int src = kk * Dm + n;
  tq[i] = (bf16x)Wq[src];
  tk[i] = (bf16x)Wk[src];
  tv[i] = (bf16x)Wv[src];
  to_[i] = (bf16x)Wo[src];
}

// ---------------------------------------------------------------------------
// 3) Projection GEMM: out = X[8192,512] @ W (+bias), wave tile 32x64.
//    vTransposed==0: write [B,H,S,64]; ==1: write [B,H,64,S] (for V).
// ---------------------------------------------------------------------------
__global__ void proj_gemm_kernel(const bf16x* __restrict__ X,
                                 const bf16x* __restrict__ WT,
                                 const float* __restrict__ bias,
                                 bf16x* __restrict__ out, int vTransposed) {
  const int lane = threadIdx.x & 31;
  const int row  = lane & 15;
  const int hi   = lane >> 4;
  const int wave = (int)(blockIdx.x * (blockDim.x >> 5) + (threadIdx.x >> 5));
  const int nTiles = Dm / 64;            // 8
  const int mBase = (wave / nTiles) * 32;
  const int nBase = (wave % nTiles) * 64;

  v8f acc[2][4];
#pragma unroll
  for (int m = 0; m < 2; ++m)
#pragma unroll
    for (int n = 0; n < 4; ++n) acc[m][n] = vzero8();

  for (int k0 = 0; k0 < Dm; k0 += 32) {
    Frag32B a[2], b[4];
#pragma unroll
    for (int m = 0; m < 2; ++m) {
      const bf16x* pa = X + (size_t)(mBase + m * 16 + row) * Dm + k0 + hi * 16;
      a[m].q[0] = *(const uint4*)pa;
      a[m].q[1] = *(const uint4*)(pa + 8);
    }
#pragma unroll
    for (int n = 0; n < 4; ++n) {
      const bf16x* pb = WT + (size_t)(nBase + n * 16 + row) * Dm + k0 + hi * 16;
      b[n].q[0] = *(const uint4*)pb;
      b[n].q[1] = *(const uint4*)(pb + 8);
    }
#pragma unroll
    for (int m = 0; m < 2; ++m)
#pragma unroll
      for (int n = 0; n < 4; ++n)
        acc[m][n] = wmma_bf16(a[m].v, b[n].v, acc[m][n]);
  }

#pragma unroll
  for (int m = 0; m < 2; ++m)
#pragma unroll
    for (int n = 0; n < 4; ++n)
#pragma unroll
      for (int j = 0; j < 8; ++j) {
        int gm = mBase + m * 16 + (hi ? 8 + j : j);
        int gn = nBase + n * 16 + row;
        float val = acc[m][n][j] + bias[gn];
        int bI = gm >> 11, s = gm & (Ss - 1);
        int h = gn >> 6, d = gn & (Dk - 1);
        size_t idx = vTransposed
                         ? ((((size_t)bI * Hh + h) * Dk + d) * Ss + s)
                         : ((((size_t)bI * Hh + h) * Ss + s) * Dk + d);
        out[idx] = (bf16x)val;
      }
}

// ---------------------------------------------------------------------------
// 4) Flash attention: one wave per 16-row q tile, 32 keys per step.
//    q,k: [B,H,S,64] bf16;  vT: [B,H,64,S] bf16;  attnOut: [B,S,512] bf16.
// ---------------------------------------------------------------------------
__global__ void attn_kernel(const bf16x* __restrict__ q,
                            const bf16x* __restrict__ k,
                            const bf16x* __restrict__ vT,
                            bf16x* __restrict__ attnOut) {
  __shared__ alignas(16) bf16x plds[8][16][48];  // per-wave P tile, padded rows
  const int lane = threadIdx.x & 31;
  const int row  = lane & 15;
  const int hi   = lane >> 4;
  const int w    = threadIdx.x >> 5;
  const int bh   = blockIdx.y;                   // 0..31 = b*8+h
  const int qBase = (blockIdx.x * 8 + w) * 16;   // q-tile row base in S

  const bf16x* qh = q  + (size_t)bh * Ss * Dk;
  const bf16x* kh = k  + (size_t)bh * Ss * Dk;
  const bf16x* vh = vT + (size_t)bh * Dk * Ss;

  // q A-frags for d 0..31 and 32..63 (invariant over the key loop)
  Frag32B qa[2];
#pragma unroll
  for (int t = 0; t < 2; ++t) {
    const bf16x* p = qh + (size_t)(qBase + row) * Dk + t * 32 + hi * 16;
    qa[t].q[0] = *(const uint4*)p;
    qa[t].q[1] = *(const uint4*)(p + 8);
  }

  v8f acc[4];
#pragma unroll
  for (int d = 0; d < 4; ++d) acc[d] = vzero8();
  float mrow[8], lrow[8];
#pragma unroll
  for (int j = 0; j < 8; ++j) { mrow[j] = -3.0e38f; lrow[j] = 0.0f; }

  const float scale = 0.04419417382415922f;  // 1/sqrt(512) (reference quirk)
  const int kEnd = qBase + 16;               // causal: keys < kEnd

  for (int k0 = 0; k0 < kEnd; k0 += 32) {
    // ---- scores: two 16-key tiles, contraction over d (2 x K=32) ----
    v8f sc[2] = {vzero8(), vzero8()};
#pragma unroll
    for (int t = 0; t < 2; ++t) {
      int kr = k0 + t * 16 + row;
      if (kr > Ss - 1) kr = Ss - 1;            // masked lanes only
#pragma unroll
      for (int dblk = 0; dblk < 2; ++dblk) {
        Frag32B kb;
        const bf16x* p = kh + (size_t)kr * Dk + dblk * 32 + hi * 16;
        kb.q[0] = *(const uint4*)p;
        kb.q[1] = *(const uint4*)(p + 8);
        sc[t] = wmma_bf16(qa[dblk].v, kb.v, sc[t]);
      }
    }

    // ---- online softmax (per-row state; row i in 16-lane half 'hi') ----
    float alpha[8], p0[8], p1[8];
#pragma unroll
    for (int j = 0; j < 8; ++j) {
      int gm  = qBase + (hi ? 8 + j : j);
      int gn0 = k0 + row;
      int gn1 = k0 + 16 + row;
      float s0 = (gn0 <= gm) ? sc[0][j] * scale : -3.0e38f;
      float s1 = (gn1 <= gm) ? sc[1][j] * scale : -3.0e38f;
      float c = fmaxf(s0, s1);
#pragma unroll
      for (int off = 1; off < 16; off <<= 1)
        c = fmaxf(c, __shfl_xor(c, off, 16));
      float mn = fmaxf(mrow[j], c);
      float a  = __expf(mrow[j] - mn);
      float e0 = (gn0 <= gm) ? __expf(s0 - mn) : 0.0f;
      float e1 = (gn1 <= gm) ? __expf(s1 - mn) : 0.0f;
      float ps = e0 + e1;
#pragma unroll
      for (int off = 1; off < 16; off <<= 1)
        ps += __shfl_xor(ps, off, 16);
      lrow[j] = lrow[j] * a + ps;
      mrow[j] = mn;
      alpha[j] = a;
      p0[j] = e0;
      p1[j] = e1;
    }
#pragma unroll
    for (int d = 0; d < 4; ++d)
#pragma unroll
      for (int j = 0; j < 8; ++j) acc[d][j] *= alpha[j];

    // ---- stage P (D-layout) into LDS, reload in A-layout ----
#pragma unroll
    for (int j = 0; j < 8; ++j) {
      int r = hi ? 8 + j : j;
      plds[w][r][row]      = (bf16x)p0[j];
      plds[w][r][16 + row] = (bf16x)p1[j];
    }
    Frag32B pa;
    {
      const bf16x* p = &plds[w][row][hi * 16];
      pa.q[0] = *(const uint4*)p;
      pa.q[1] = *(const uint4*)(p + 8);
    }

    // ---- acc += P(16x32) @ V(32keys x 64d), vT gives contiguous B loads ----
    int kc = k0 + hi * 16;
    if (kc > Ss - 16) kc = Ss - 16;            // masked keys only
#pragma unroll
    for (int d = 0; d < 4; ++d) {
      Frag32B vb;
      const bf16x* p = vh + (size_t)(d * 16 + row) * Ss + kc;
      vb.q[0] = *(const uint4*)p;
      vb.q[1] = *(const uint4*)(p + 8);
      acc[d] = wmma_bf16(pa.v, vb.v, acc[d]);
    }
  }

  // ---- finalize: out = acc / l, write concat layout [B,S,H*64] ----
  const int bI = bh >> 3, h = bh & 7;
#pragma unroll
  for (int d = 0; d < 4; ++d)
#pragma unroll
    for (int j = 0; j < 8; ++j) {
      int gm = qBase + (hi ? 8 + j : j);
      int gd = d * 16 + row;
      float o = acc[d][j] / lrow[j];
      size_t idx = ((size_t)bI * Ss + gm) * Dm + h * Dk + gd;
      attnOut[idx] = (bf16x)o;
    }
}

// ---------------------------------------------------------------------------
// 5) Output projection: attnOut[8192,512] @ Wo + bo -> fp32 d_out
// ---------------------------------------------------------------------------
__global__ void out_gemm_kernel(const bf16x* __restrict__ X,
                                const bf16x* __restrict__ WT,
                                const float* __restrict__ bias,
                                float* __restrict__ out) {
  const int lane = threadIdx.x & 31;
  const int row  = lane & 15;
  const int hi   = lane >> 4;
  const int wave = (int)(blockIdx.x * (blockDim.x >> 5) + (threadIdx.x >> 5));
  const int nTiles = Dm / 64;
  const int mBase = (wave / nTiles) * 32;
  const int nBase = (wave % nTiles) * 64;

  v8f acc[2][4];
#pragma unroll
  for (int m = 0; m < 2; ++m)
#pragma unroll
    for (int n = 0; n < 4; ++n) acc[m][n] = vzero8();

  for (int k0 = 0; k0 < Dm; k0 += 32) {
    Frag32B a[2], b[4];
#pragma unroll
    for (int m = 0; m < 2; ++m) {
      const bf16x* pa = X + (size_t)(mBase + m * 16 + row) * Dm + k0 + hi * 16;
      a[m].q[0] = *(const uint4*)pa;
      a[m].q[1] = *(const uint4*)(pa + 8);
    }
#pragma unroll
    for (int n = 0; n < 4; ++n) {
      const bf16x* pb = WT + (size_t)(nBase + n * 16 + row) * Dm + k0 + hi * 16;
      b[n].q[0] = *(const uint4*)pb;
      b[n].q[1] = *(const uint4*)(pb + 8);
    }
#pragma unroll
    for (int m = 0; m < 2; ++m)
#pragma unroll
      for (int n = 0; n < 4; ++n)
        acc[m][n] = wmma_bf16(a[m].v, b[n].v, acc[m][n]);
  }

#pragma unroll
  for (int m = 0; m < 2; ++m)
#pragma unroll
    for (int n = 0; n < 4; ++n)
#pragma unroll
      for (int j = 0; j < 8; ++j) {
        int gm = mBase + m * 16 + (hi ? 8 + j : j);
        int gn = nBase + n * 16 + row;
        out[(size_t)gm * Dm + gn] = acc[m][n][j] + bias[gn];
      }
}

// ---------------------------------------------------------------------------
extern "C" void kernel_launch(void* const* d_in, const int* in_sizes, int n_in,
                              void* d_out, int out_size, void* d_ws, size_t ws_size,
                              hipStream_t stream) {
  const float* Q  = (const float*)d_in[0];
  const float* K  = (const float*)d_in[1];
  const float* V  = (const float*)d_in[2];
  const float* Wq = (const float*)d_in[3];
  const float* bq = (const float*)d_in[4];
  const float* Wk = (const float*)d_in[5];
  const float* bk = (const float*)d_in[6];
  const float* Wv = (const float*)d_in[7];
  const float* bv = (const float*)d_in[8];
  const float* Wo = (const float*)d_in[9];
  const float* bo = (const float*)d_in[10];
  float* out = (float*)d_out;

  char* ws = (char*)d_ws;
  size_t off = 0;
  auto carve = [&](size_t bytes) -> char* {
    char* p = ws + off;
    off += (bytes + 4095) & ~(size_t)4095;  // pad between regions (OOB slack)
    return p;
  };
  const size_t actBytes = (size_t)Mrows * Dm * sizeof(bf16x);  // 8.39 MB
  const size_t wBytes   = (size_t)Dm * Dm * sizeof(bf16x);     // 0.52 MB

  bf16x* Xq  = (bf16x*)carve(actBytes);
  bf16x* Xk  = (bf16x*)carve(actBytes);
  bf16x* Xv  = (bf16x*)carve(actBytes);
  bf16x* WTq = (bf16x*)carve(wBytes);
  bf16x* WTk = (bf16x*)carve(wBytes);
  bf16x* WTv = (bf16x*)carve(wBytes);
  bf16x* WTo = (bf16x*)carve(wBytes);
  bf16x* qb  = (bf16x*)carve(actBytes);   // [B,H,S,64]
  bf16x* kb  = (bf16x*)carve(actBytes);   // [B,H,S,64]
  bf16x* vTb = (bf16x*)carve(actBytes);   // [B,H,64,S]
  bf16x* ao  = (bf16x*)carve(actBytes);   // [B,S,512]

  // 1) activations -> bf16
  {
    size_t n = (size_t)Mrows * Dm;
    int blocks = (int)((n + 255) / 256);
    cvt_qkv_kernel<<<blocks, 256, 0, stream>>>(Q, K, V, Xq, Xk, Xv);
  }
  // 2) weights -> transposed bf16
  cvt_w_kernel<<<(Dm * Dm) / 256, 256, 0, stream>>>(Wq, Wk, Wv, Wo,
                                                    WTq, WTk, WTv, WTo);
  // 3) projections (256 blocks x 8 waves = 2048 tile-waves of 32x64)
  proj_gemm_kernel<<<256, 256, 0, stream>>>(Xq, WTq, bq, qb, 0);
  proj_gemm_kernel<<<256, 256, 0, stream>>>(Xk, WTk, bk, kb, 0);
  proj_gemm_kernel<<<256, 256, 0, stream>>>(Xv, WTv, bv, vTb, 1);
  // 4) flash attention: 128 q-tiles per (b,h), 8 waves/block
  attn_kernel<<<dim3(16, Bb * Hh), 256, 0, stream>>>(qb, kb, vTb, ao);
  // 5) output projection -> fp32
  out_gemm_kernel<<<256, 256, 0, stream>>>(ao, WTo, bo, out);
}
